// S3MNet_46411416601227
// MI455X (gfx1250) — matrix operand order repaired and checked by reading, not compiled, not run on hardware
//
#include <hip/hip_runtime.h>

#define N_NODES 80000
#define N_EDGES 1280000
#define BATCH   4
#define NV      20000
#define K_EIG   50
#define FEAT    256

typedef float v8f  __attribute__((ext_vector_type(8)));
typedef __bf16 v16bf __attribute__((ext_vector_type(16)));

__device__ __forceinline__ unsigned f2bf_u(float f) {
    union { float f; unsigned u; } x; x.f = f;
    return (x.u + 0x7FFFu + ((x.u >> 16) & 1u)) >> 16;   // round-to-nearest-even
}
__device__ __forceinline__ __bf16 f2bf(float f) {
    union { unsigned short s; __bf16 b; } y;
    y.s = (unsigned short)f2bf_u(f);
    return y.b;
}
__device__ __forceinline__ unsigned pack2bf(float a, float b) {
    return f2bf_u(a) | (f2bf_u(b) << 16);
}

// ---------------------------------------------------------------- utilities
__global__ void zero_f32(float* __restrict__ p, long long n) {
    long long i = (long long)blockIdx.x * blockDim.x + threadIdx.x;
    long long stride = (long long)gridDim.x * blockDim.x;
    for (; i < n; i += stride) p[i] = 0.0f;
}

__global__ void deg_kernel(const int* __restrict__ ei, const float* __restrict__ ew,
                           float* __restrict__ deg, int E) {
    int e = blockIdx.x * blockDim.x + threadIdx.x;
    if (e >= E) return;
    atomicAdd(&deg[ei[E + e]], ew[e]);
}

__global__ void norm_kernel(const int* __restrict__ ei, const float* __restrict__ ew,
                            const float* __restrict__ deg, float* __restrict__ nrm, int E) {
    int e = blockIdx.x * blockDim.x + threadIdx.x;
    if (e >= E) return;
    int s = ei[e], d = ei[E + e];
    float ds = deg[s], dd = deg[d];
    float is = ds > 0.f ? rsqrtf(ds) : 0.f;
    float id = dd > 0.f ? rsqrtf(dd) : 0.f;
    nrm[e] = is * ew[e] * id;
}

// out[dst] += nrm[e] * h[src]  (F multiple of 4, chunk of 4 per blockIdx.y)
__global__ void prop4_kernel(const int* __restrict__ ei, const float* __restrict__ nrm,
                             const float* __restrict__ h, float* __restrict__ out,
                             int E, int F) {
    int e = blockIdx.x * blockDim.x + threadIdx.x;
    if (e >= E) return;
    int s = ei[e], d = ei[E + e];
    float w = nrm[e];
    int c4 = blockIdx.y * 4;
    const float4 v = *(const float4*)(h + (size_t)s * F + c4);
    float* op = out + (size_t)d * F + c4;
    atomicAdd(op + 0, w * v.x);
    atomicAdd(op + 1, w * v.y);
    atomicAdd(op + 2, w * v.z);
    atomicAdd(op + 3, w * v.w);
}

// scalar variant for small F (layer-1 input has F=3)
__global__ void prop1_kernel(const int* __restrict__ ei, const float* __restrict__ nrm,
                             const float* __restrict__ h, float* __restrict__ out,
                             int E, int F) {
    int e = blockIdx.x * blockDim.x + threadIdx.x;
    if (e >= E) return;
    int s = ei[e], d = ei[E + e];
    float w = nrm[e];
    for (int f = 0; f < F; ++f)
        atomicAdd(out + (size_t)d * F + f, w * h[(size_t)s * F + f]);
}

// ---------------------------------------------------------------- WMMA GEMM
// C[M,N] (+)= A[M,K] * W  (+ bias, relu).  W is [K,N] row-major, or [N,K] if transB.
// 64x64 block tile, 4 waves, each wave does a 16-row stripe, 4 wmma tiles wide.
// aligned!=0 => M%64==0, K%32==0, N%64==0, transB==0: vectorized guard-free fill.
#define TILE  64
#define KSTEP 32
#define LDK   40   // padded K pitch in LDS (bank-conflict avoidance)

__global__ void __launch_bounds__(128)
gemm_wmma(const float* __restrict__ A, const float* __restrict__ W,
          const float* __restrict__ bias, float* __restrict__ C,
          int M, int K, int N, int transB, int acc, int relu, int aligned,
          long long sA, long long sW, long long sC) {
    A += (long long)blockIdx.z * sA;
    W += (long long)blockIdx.z * sW;
    C += (long long)blockIdx.z * sC;

    __shared__ __bf16 As[TILE * LDK];   // [row][k]
    __shared__ __bf16 Bs[TILE * LDK];   // [col][k]  (stored K-contiguous)

    const int tid  = threadIdx.x;
    const int wave = tid >> 5;
    const int lane = tid & 31;
    const int rowBase = blockIdx.y * TILE;
    const int colBase = blockIdx.x * TILE;

    v8f cacc[4];
    const v8f z8 = {0.f, 0.f, 0.f, 0.f, 0.f, 0.f, 0.f, 0.f};
    for (int t = 0; t < 4; ++t) cacc[t] = z8;

    for (int k0 = 0; k0 < K; k0 += KSTEP) {
        if (aligned) {
            // A tile: float4 loads, packed 2xbf16 dword LDS stores (no guards)
            #pragma unroll
            for (int i = tid; i < TILE * KSTEP / 4; i += 128) {
                int r  = i >> 3;          // 0..63
                int k4 = (i & 7) << 2;    // 0,4,...,28
                const float4 v = *(const float4*)(A + (size_t)(rowBase + r) * K + (k0 + k4));
                unsigned* dst = (unsigned*)&As[r * LDK + k4];
                dst[0] = pack2bf(v.x, v.y);
                dst[1] = pack2bf(v.z, v.w);
            }
            // B tile: W[k*N+n] -> Bs[n][k] (transposed store, scalar but guard-free)
            #pragma unroll
            for (int i = tid; i < TILE * KSTEP; i += 128) {
                int n = i >> 5, k = i & 31;
                Bs[n * LDK + k] = f2bf(W[(size_t)(k0 + k) * N + (colBase + n)]);
            }
            // prefetch next A K-tile while WMMA runs (global_prefetch_b8)
            if (k0 + KSTEP < K) {
                int r = tid >> 1;
                __builtin_prefetch(A + (size_t)(rowBase + r) * K + (k0 + KSTEP) + (tid & 1) * 16,
                                   0, 1);
            }
        } else {
            for (int i = tid; i < TILE * KSTEP; i += 128) {
                int r  = i >> 5;
                int k  = i & 31;
                int gk = k0 + k;
                float av = 0.f, wv = 0.f;
                int gr = rowBase + r;
                if (gr < M && gk < K) av = A[(size_t)gr * K + gk];
                int gn = colBase + r;
                if (gn < N && gk < K)
                    wv = transB ? W[(size_t)gn * K + gk] : W[(size_t)gk * N + gn];
                As[r * LDK + k] = f2bf(av);
                Bs[r * LDK + k] = f2bf(wv);
            }
        }
        __syncthreads();

        // A fragment (16x32 bf16): lanes 0-15 -> K {0..7,16..23}, lanes 16-31 -> K {8..15,24..31}
        const int arow = wave * 16 + (lane & 15);
        const int akb  = (lane >= 16) ? 8 : 0;
        const __bf16* ap = &As[arow * LDK];
        v16bf afrag;
        #pragma unroll
        for (int t = 0; t < 8; ++t) {
            afrag[t]     = ap[akb + t];
            afrag[8 + t] = ap[16 + akb + t];
        }
        // B fragments (32x16 bf16): lanes 0-15 -> K {0..7,16..23}, lanes 16-31 -> K {8..15,24..31}
        const int bkb = (lane >= 16) ? 8 : 0;
        #pragma unroll
        for (int ct = 0; ct < 4; ++ct) {
            const __bf16* bp = &Bs[(ct * 16 + (lane & 15)) * LDK + bkb];
            v16bf bfrag;
            #pragma unroll
            for (int t = 0; t < 8; ++t) { bfrag[t] = bp[t]; bfrag[8 + t] = bp[16 + t]; }
            cacc[ct] = __builtin_amdgcn_wmma_f32_16x16x32_bf16(
                false, afrag, false, bfrag, (short)0, cacc[ct], false, false);
        }
        __syncthreads();
    }

    // epilogue: C/D layout — lanes 0-15: rows 0..7, lanes 16-31: rows 8..15
    const int rb = rowBase + wave * 16 + ((lane >= 16) ? 8 : 0);
    #pragma unroll
    for (int ct = 0; ct < 4; ++ct) {
        int n = colBase + ct * 16 + (lane & 15);
        if (n >= N) continue;
        float bv = bias ? bias[n] : 0.f;
        #pragma unroll
        for (int r = 0; r < 8; ++r) {
            int row = rb + r;
            if (row >= M) continue;
            size_t idx = (size_t)row * N + n;
            float v = cacc[ct][r];
            if (acc) v += C[idx];
            v += bv;
            if (relu) v = fmaxf(v, 0.f);
            C[idx] = v;
        }
    }
}

// ---------------------------------------------------------------- L2 norm rows (FEAT=256)
__global__ void l2norm_rows(const float* __restrict__ in, float* __restrict__ out, int rows) {
    int row = blockIdx.x;
    if (row >= rows) return;
    __shared__ float sred[64];
    int t = threadIdx.x;                       // 64 threads, 4 floats each
    const float4 v = ((const float4*)(in + (size_t)row * FEAT))[t];
    sred[t] = v.x * v.x + v.y * v.y + v.z * v.z + v.w * v.w;
    __syncthreads();
    for (int off = 32; off > 0; off >>= 1) {
        if (t < off) sred[t] += sred[t + off];
        __syncthreads();
    }
    float scale = 1.0f / fmaxf(sqrtf(sred[0]), 1e-12f);
    float4 o; o.x = v.x * scale; o.y = v.y * scale; o.z = v.z * scale; o.w = v.w * scale;
    ((float4*)(out + (size_t)row * FEAT))[t] = o;
}

// ---------------------------------------------------------------- 50x50 Gauss-Jordan per batch
// Solves A X = B, writes out = X^T  (reference: transpose(solve(A,B)))
__global__ void solve_gje(const float* __restrict__ Ag, const float* __restrict__ Bg,
                          float* __restrict__ outT) {
    int b = blockIdx.x;
    __shared__ float A[K_EIG][K_EIG + 1];
    __shared__ float Bm[K_EIG][K_EIG + 1];
    __shared__ float fac[K_EIG];
    int tid = threadIdx.x;
    for (int i = tid; i < K_EIG * K_EIG; i += blockDim.x) {
        int r = i / K_EIG, c = i % K_EIG;
        A[r][c]  = Ag[(size_t)b * K_EIG * K_EIG + i];
        Bm[r][c] = Bg[(size_t)b * K_EIG * K_EIG + i];
    }
    __syncthreads();
    for (int p = 0; p < K_EIG; ++p) {
        if (tid < K_EIG) fac[tid] = (tid == p) ? 0.f : A[tid][p] / A[p][p];
        __syncthreads();
        for (int i = tid; i < K_EIG * K_EIG; i += blockDim.x) {
            int r = i / K_EIG, c = i % K_EIG;
            if (r != p) {
                A[r][c]  -= fac[r] * A[p][c];
                Bm[r][c] -= fac[r] * Bm[p][c];
            }
        }
        __syncthreads();
    }
    for (int i = tid; i < K_EIG * K_EIG; i += blockDim.x) {
        int r = i / K_EIG, c = i % K_EIG;
        // X[r][c] = Bm[r][c]/A[r][r];  out[c][r] = X[r][c]
        outT[(size_t)b * K_EIG * K_EIG + (size_t)c * K_EIG + r] = Bm[r][c] / A[r][r];
    }
}

// ---------------------------------------------------------------- host-side helpers
static inline void launch_zero(float* p, long long n, hipStream_t st) {
    int blocks = (int)(((n + 255) / 256) < 2048 ? ((n + 255) / 256) : 2048);
    zero_f32<<<blocks, 256, 0, st>>>(p, n);
}

static inline void launch_gemm(const float* A, const float* W, const float* bias, float* C,
                               int M, int K, int N, int transB, int acc, int relu,
                               long long sA, long long sW, long long sC, int batch,
                               hipStream_t st) {
    int aligned = (M % TILE == 0 && K % KSTEP == 0 && N % TILE == 0 && !transB) ? 1 : 0;
    dim3 g((N + TILE - 1) / TILE, (M + TILE - 1) / TILE, batch);
    gemm_wmma<<<g, 128, 0, st>>>(A, W, bias, C, M, K, N, transB, acc, relu, aligned,
                                 sA, sW, sC);
}

struct GraphWeights {
    const float *W1_0, *W1_1, *b1;
    const float *W2_0, *W2_1, *W2_2, *b2;
    const float *W3_0, *W3_1, *W3_2, *W3_3, *b3;
    const float *Wfc, *bfc;
};

static void process_graph(const float* x, const int* ei, const float* ea,
                          const GraphWeights& w, float* feat_out,
                          float* bufA, float* bufB, float* bufC,
                          float* deg, float* enrm, hipStream_t st) {
    const int N = N_NODES, E = N_EDGES;
    const int eb = (E + 255) / 256;

    // normalization coefficients
    launch_zero(deg, N, st);
    deg_kernel<<<eb, 256, 0, st>>>(ei, ea, deg, E);
    norm_kernel<<<eb, 256, 0, st>>>(ei, ea, deg, enrm, E);

    // ---- TAGConv layer 1: 3 -> 64, K=1
    launch_gemm(x, w.W1_0, w.b1, bufA, N, 3, 64, 0, 0, 0, 0, 0, 0, 1, st);
    launch_zero(bufB, (long long)N * 3, st);
    prop1_kernel<<<eb, 256, 0, st>>>(ei, enrm, x, bufB, E, 3);
    launch_gemm(bufB, w.W1_1, nullptr, bufA, N, 3, 64, 0, 1, 1, 0, 0, 0, 1, st);
    // h1 = bufA [N,64]

    // ---- TAGConv layer 2: 64 -> 128, K=2
    launch_gemm(bufA, w.W2_0, w.b2, bufC, N, 64, 128, 0, 0, 0, 0, 0, 0, 1, st);
    launch_zero(bufB, (long long)N * 64, st);
    prop4_kernel<<<dim3(eb, 16), 256, 0, st>>>(ei, enrm, bufA, bufB, E, 64);
    launch_gemm(bufB, w.W2_1, nullptr, bufC, N, 64, 128, 0, 1, 0, 0, 0, 0, 1, st);
    launch_zero(bufA, (long long)N * 64, st);
    prop4_kernel<<<dim3(eb, 16), 256, 0, st>>>(ei, enrm, bufB, bufA, E, 64);
    launch_gemm(bufA, w.W2_2, nullptr, bufC, N, 64, 128, 0, 1, 1, 0, 0, 0, 1, st);
    // h2 = bufC [N,128]

    // ---- TAGConv layer 3: 128 -> 256, K=3
    launch_gemm(bufC, w.W3_0, w.b3, bufA, N, 128, 256, 0, 0, 0, 0, 0, 0, 1, st);
    launch_zero(bufB, (long long)N * 128, st);
    prop4_kernel<<<dim3(eb, 32), 256, 0, st>>>(ei, enrm, bufC, bufB, E, 128);
    launch_gemm(bufB, w.W3_1, nullptr, bufA, N, 128, 256, 0, 1, 0, 0, 0, 0, 1, st);
    launch_zero(bufC, (long long)N * 128, st);
    prop4_kernel<<<dim3(eb, 32), 256, 0, st>>>(ei, enrm, bufB, bufC, E, 128);
    launch_gemm(bufC, w.W3_2, nullptr, bufA, N, 128, 256, 0, 1, 0, 0, 0, 0, 1, st);
    launch_zero(bufB, (long long)N * 128, st);
    prop4_kernel<<<dim3(eb, 32), 256, 0, st>>>(ei, enrm, bufC, bufB, E, 128);
    launch_gemm(bufB, w.W3_3, nullptr, bufA, N, 128, 256, 0, 1, 1, 0, 0, 0, 1, st);
    // h3 = bufA [N,256]

    // ---- FC + row L2 normalize -> feat_out (in d_out)
    launch_gemm(bufA, w.Wfc, w.bfc, bufB, N, 256, 256, 0, 0, 0, 0, 0, 0, 1, st);
    l2norm_rows<<<N, 64, 0, st>>>(bufB, feat_out, N);
}

extern "C" void kernel_launch(void* const* d_in, const int* in_sizes, int n_in,
                              void* d_out, int out_size, void* d_ws, size_t ws_size,
                              hipStream_t stream) {
    (void)in_sizes; (void)n_in; (void)out_size; (void)ws_size;

    const float* x_x  = (const float*)d_in[0];
    const int*   ei_x = (const int*)  d_in[1];
    const float* ea_x = (const float*)d_in[2];
    const float* x_y  = (const float*)d_in[3];
    const int*   ei_y = (const int*)  d_in[4];
    const float* ea_y = (const float*)d_in[5];
    const float* evx  = (const float*)d_in[6];   // [B, K_EIG, NV]
    const float* evy  = (const float*)d_in[7];

    GraphWeights w;
    w.W1_0 = (const float*)d_in[8];  w.W1_1 = (const float*)d_in[9];  w.b1 = (const float*)d_in[10];
    w.W2_0 = (const float*)d_in[11]; w.W2_1 = (const float*)d_in[12]; w.W2_2 = (const float*)d_in[13];
    w.b2   = (const float*)d_in[14];
    w.W3_0 = (const float*)d_in[15]; w.W3_1 = (const float*)d_in[16]; w.W3_2 = (const float*)d_in[17];
    w.W3_3 = (const float*)d_in[18]; w.b3   = (const float*)d_in[19];
    w.Wfc  = (const float*)d_in[20]; w.bfc  = (const float*)d_in[21];

    float* out    = (float*)d_out;
    float* c_xy   = out;                                   // [B,50,50]
    float* c_yx   = out + BATCH * K_EIG * K_EIG;           // [B,50,50]
    float* feat_x = out + 2 * BATCH * K_EIG * K_EIG;       // [B,NV,FEAT] == [N_NODES,FEAT]
    float* feat_y = feat_x + (size_t)N_NODES * FEAT;

    // workspace carve-out
    char* ws = (char*)d_ws;
    size_t off = 0;
    auto carve = [&](size_t bytes) -> float* {
        float* p = (float*)(ws + off);
        off += (bytes + 255) & ~(size_t)255;
        return p;
    };
    float* bufA = carve((size_t)N_NODES * 256 * sizeof(float));
    float* bufB = carve((size_t)N_NODES * 256 * sizeof(float));
    float* bufC = carve((size_t)N_NODES * 128 * sizeof(float));
    float* deg  = carve((size_t)N_NODES * sizeof(float));
    float* enrm = carve((size_t)N_EDGES * sizeof(float));
    float* fhx  = carve((size_t)BATCH * K_EIG * FEAT * sizeof(float));
    float* fhy  = carve((size_t)BATCH * K_EIG * FEAT * sizeof(float));
    float* Aff  = carve((size_t)BATCH * K_EIG * K_EIG * sizeof(float));
    float* Agg  = carve((size_t)BATCH * K_EIG * K_EIG * sizeof(float));
    float* Bfg  = carve((size_t)BATCH * K_EIG * K_EIG * sizeof(float));
    float* Bgf  = carve((size_t)BATCH * K_EIG * K_EIG * sizeof(float));

    // feature extraction for both graphs (writes feats into d_out)
    process_graph(x_x, ei_x, ea_x, w, feat_x, bufA, bufB, bufC, deg, enrm, stream);
    process_graph(x_y, ei_y, ea_y, w, feat_y, bufA, bufB, bufC, deg, enrm, stream);

    // spectral projections: f_hat[b] = E_b(50 x 20000) * feat_b(20000 x 256), batched over z
    launch_gemm(evx, feat_x, nullptr, fhx, K_EIG, NV, FEAT, 0, 0, 0,
                (long long)K_EIG * NV, (long long)NV * FEAT, (long long)K_EIG * FEAT,
                BATCH, stream);
    launch_gemm(evy, feat_y, nullptr, fhy, K_EIG, NV, FEAT, 0, 0, 0,
                (long long)K_EIG * NV, (long long)NV * FEAT, (long long)K_EIG * FEAT,
                BATCH, stream);

    // Gram matrices: X * Y^T  (transB path), batched over z
    const long long sF = (long long)K_EIG * FEAT, sG = (long long)K_EIG * K_EIG;
    launch_gemm(fhx, fhx, nullptr, Aff, K_EIG, FEAT, K_EIG, 1, 0, 0, sF, sF, sG, BATCH, stream);
    launch_gemm(fhy, fhy, nullptr, Agg, K_EIG, FEAT, K_EIG, 1, 0, 0, sF, sF, sG, BATCH, stream);
    launch_gemm(fhx, fhy, nullptr, Bfg, K_EIG, FEAT, K_EIG, 1, 0, 0, sF, sF, sG, BATCH, stream);
    launch_gemm(fhy, fhx, nullptr, Bgf, K_EIG, FEAT, K_EIG, 1, 0, 0, sF, sF, sG, BATCH, stream);

    // c_xy = solve(A_ff, B_fg)^T ; c_yx = solve(A_gg, B_gf)^T
    solve_gje<<<BATCH, 128, 0, stream>>>(Aff, Bfg, c_xy);
    solve_gje<<<BATCH, 128, 0, stream>>>(Agg, Bgf, c_yx);
}
